// InterSelfAttention_3135326126158
// MI455X (gfx1250) — matrix operand order repaired and checked by reading, not compiled, not run on hardware
//
#include <hip/hip_runtime.h>

typedef __attribute__((ext_vector_type(16))) _Float16 v16h;
typedef __attribute__((ext_vector_type(8)))  _Float16 v8h;
typedef __attribute__((ext_vector_type(8)))  float    v8f;
typedef __attribute__((ext_vector_type(4)))  unsigned int u32x4;
typedef __attribute__((ext_vector_type(8)))  int      i32x8;
typedef __attribute__((ext_vector_type(4)))  int      i32x4;

#define B_  8
#define H_  16
#define S_  512
#define DH_ 64
#define HID_ 1024
#define LDT 40   // LDS row stride in halfs (80B, 16B-aligned rows)

// ---------------------------------------------------------------------------
// TDM: DMA a 2D f16 tile (rows x 32 halfs, global row stride in halfs) into
// LDS with padding 16 DWORDs data + 4 DWORDs pad  => LDS row stride 40 halfs.
// D# layout per CDNA5 ISA ch.8 (group0 128b, group1 256b; groups 2/3 unused).
// ---------------------------------------------------------------------------
__device__ __forceinline__ void tdm_load_tile_f16(const _Float16* gsrc,
                                                  unsigned lds_byte_addr,
                                                  unsigned rows,
                                                  unsigned long long row_stride_halfs) {
  unsigned long long ga = (unsigned long long)(size_t)gsrc;
  u32x4 g0;
  g0[0] = 1u;                                        // count=1, user mode
  g0[1] = lds_byte_addr;                             // lds_addr
  g0[2] = (unsigned)(ga & 0xFFFFFFFFu);              // global_addr[31:0]
  g0[3] = (unsigned)((ga >> 32) & 0x01FFFFFFu) | (2u << 30); // addr[56:32] | type=2

  const unsigned dim0 = 1u << 20, dim1 = 1u << 20;   // large => no OOB clipping
  const unsigned tile0 = 32;                         // halfs per row
  unsigned long long s0 = row_stride_halfs;          // data_size units
  unsigned long long s1 = 0;
  i32x8 g1;
  // data_size=1 (2B) | pad_enable | pad_interval=3 (16 DW) | pad_amount=3 (4 DW)
  g1[0] = (int)((1u << 16) | (1u << 20) | (3u << 22) | (3u << 25));
  g1[1] = (int)((dim0 & 0xFFFFu) << 16);                       // tensor_dim0[15:0]
  g1[2] = (int)((dim0 >> 16) | ((dim1 & 0xFFFFu) << 16));      // dim0 hi | dim1 lo
  g1[3] = (int)((dim1 >> 16) | (tile0 << 16));                 // dim1 hi | tile_dim0
  g1[4] = (int)(rows & 0xFFFFu);                               // tile_dim1 (tile_dim2=0)
  g1[5] = (int)(s0 & 0xFFFFFFFFull);                           // stride0[31:0]
  g1[6] = (int)(((s0 >> 32) & 0xFFFFull) | ((s1 & 0xFFFFull) << 16));
  g1[7] = (int)((s1 >> 16) & 0xFFFFFFFFull);
  i32x4 z4 = {0, 0, 0, 0};
#if defined(__clang_major__) && (__clang_major__ >= 23)
  i32x8 z8 = {0, 0, 0, 0, 0, 0, 0, 0};
  __builtin_amdgcn_tensor_load_to_lds(g0, g1, z4, z4, z8, 0);
#else
  __builtin_amdgcn_tensor_load_to_lds(g0, g1, z4, z4, 0);
#endif
}

// ---------------------------------------------------------------------------
// lens: first-zero index of monotone 1..10..0 padding vectors == popcount
// ---------------------------------------------------------------------------
__global__ void lens_kernel(const int* __restrict__ pq, const int* __restrict__ pk,
                            int* __restrict__ lens) {
  __shared__ int acc[16];
  if (threadIdx.x < 16) acc[threadIdx.x] = 0;
  __syncthreads();
  for (int b = 0; b < B_; ++b) {
    int s0 = 0, s1 = 0;
    for (int s = threadIdx.x; s < S_; s += 256) {
      s0 += (pq[b * S_ + s] != 0);
      s1 += (pk[b * S_ + s] != 0);
    }
    atomicAdd(&acc[b], s0);
    atomicAdd(&acc[8 + b], s1);
  }
  __syncthreads();
  if (threadIdx.x < 16) lens[threadIdx.x] = acc[threadIdx.x];
}

// ---------------------------------------------------------------------------
// Fragment helpers: ISA 7.12.2 layouts for 16-bit A (16x32) and B (32x16)
// ---------------------------------------------------------------------------
__device__ __forceinline__ v16h load_a_frag(const _Float16* rowbase, int lane) {
  const int kb = (lane & 16) ? 8 : 0;
  v8h lo = *(const v8h*)(rowbase + kb);
  v8h hi = *(const v8h*)(rowbase + kb + 16);
  v16h r;
#pragma unroll
  for (int j = 0; j < 8; ++j) { r[j] = lo[j]; r[j + 8] = hi[j]; }
  return r;
}

__device__ __forceinline__ v16h load_b_frag(const _Float16* colbase, int lane) {
  const int kb = (lane & 16) ? 8 : 0;
  v8h lo = *(const v8h*)(colbase + kb);
  v8h hi = *(const v8h*)(colbase + kb + 16);
  v16h r;
#pragma unroll
  for (int j = 0; j < 8; ++j) { r[j] = lo[j]; r[j + 8] = hi[j]; }
  return r;
}

// ---------------------------------------------------------------------------
// Projection GEMM: C[m,n] = sum_k A4perm[m,k] * W[k,n] + bias[n]
//   (fp32 sources converted to f16 at staging -> manual staging path)
// ---------------------------------------------------------------------------
__global__ __launch_bounds__(256) void gemm_proj(const float* __restrict__ A4,
                                                 const float* __restrict__ W,
                                                 const float* __restrict__ bias,
                                                 _Float16* __restrict__ Cp) {
  __shared__ __attribute__((aligned(16))) _Float16 As[128 * LDT];
  __shared__ __attribute__((aligned(16))) _Float16 Bs[128 * LDT];
  const int m0 = blockIdx.x * 128, n0 = blockIdx.y * 128;
  const int tid = threadIdx.x, wid = tid >> 5, lane = tid & 31;
  const int wm = (wid & 3) * 32, wn = (wid >> 2) * 64;
  v8f acc[2][4];
#pragma unroll
  for (int i = 0; i < 2; ++i)
#pragma unroll
    for (int j = 0; j < 4; ++j)
#pragma unroll
      for (int r = 0; r < 8; ++r) acc[i][j][r] = 0.f;

  for (int k0 = 0; k0 < HID_; k0 += 32) {
    __syncthreads();
    const int hh = k0 >> 6, dd0 = k0 & 63;
#pragma unroll
    for (int i = 0; i < 16; ++i) {
      int e = tid + 256 * i, r = e >> 5, c = e & 31;
      int m = m0 + r, b = m >> 9, s = m & 511;
      As[r * LDT + c] = (_Float16)A4[(((b * H_ + hh) * S_) + s) * DH_ + dd0 + c];
    }
#pragma unroll
    for (int i = 0; i < 16; ++i) {
      int e = tid + 256 * i, n = e & 127, k = e >> 7;
      Bs[n * LDT + k] = (_Float16)W[(k0 + k) * HID_ + n0 + n];
    }
    __syncthreads();
    const int mrow = lane & 15;
    v16h af[2], bf[4];
#pragma unroll
    for (int fm = 0; fm < 2; ++fm)
      af[fm] = load_a_frag(&As[(wm + fm * 16 + mrow) * LDT], lane);
#pragma unroll
    for (int fn = 0; fn < 4; ++fn)
      bf[fn] = load_b_frag(&Bs[(wn + fn * 16 + mrow) * LDT], lane);
#pragma unroll
    for (int fm = 0; fm < 2; ++fm)
#pragma unroll
      for (int fn = 0; fn < 4; ++fn)
        acc[fm][fn] = __builtin_amdgcn_wmma_f32_16x16x32_f16(
            false, af[fm], false, bf[fn], (short)0, acc[fm][fn], false, false);
  }
  const int mr = (lane & 16) ? 8 : 0, nc = lane & 15;
#pragma unroll
  for (int fm = 0; fm < 2; ++fm)
#pragma unroll
    for (int fn = 0; fn < 4; ++fn) {
      int n = n0 + wn + fn * 16 + nc;
      float bb = bias[n];
#pragma unroll
      for (int r = 0; r < 8; ++r) {
        int m = m0 + wm + fm * 16 + mr + r;
        Cp[m * HID_ + n] = (_Float16)(acc[fm][fn][r] + bb);
      }
    }
}

// ---------------------------------------------------------------------------
// scores_t[bh][kk][qq] = 0.125 * sum_d k[kk,d]*q[qq,d]
//   f16 sources, row-contiguous: tiles staged by the Tensor Data Mover.
// ---------------------------------------------------------------------------
__global__ __launch_bounds__(256) void gemm_scores(const _Float16* __restrict__ kp,
                                                   const _Float16* __restrict__ qp,
                                                   float* __restrict__ st) {
  __shared__ __attribute__((aligned(16))) _Float16 As[128 * LDT];
  __shared__ __attribute__((aligned(16))) _Float16 Bs[128 * LDT];
  const int m0 = blockIdx.x * 128, n0 = blockIdx.y * 128, bh = blockIdx.z;
  const int b = bh >> 4, hoff = (bh & 15) * DH_;
  const int tid = threadIdx.x, wid = tid >> 5, lane = tid & 31;
  const int wm = (wid & 3) * 32, wn = (wid >> 2) * 64;
  v8f acc[2][4];
#pragma unroll
  for (int i = 0; i < 2; ++i)
#pragma unroll
    for (int j = 0; j < 4; ++j)
#pragma unroll
      for (int r = 0; r < 8; ++r) acc[i][j][r] = 0.f;

  for (int k0 = 0; k0 < DH_; k0 += 32) {
    __syncthreads();
    if (tid < 32) {   // wave 0 issues both tile DMAs, then drains TENSORcnt
      tdm_load_tile_f16(kp + (size_t)(b * S_ + m0) * HID_ + hoff + k0,
                        (unsigned)(size_t)&As[0], 128, HID_);
      tdm_load_tile_f16(qp + (size_t)(b * S_ + n0) * HID_ + hoff + k0,
                        (unsigned)(size_t)&Bs[0], 128, HID_);
      __builtin_amdgcn_s_wait_tensorcnt(0);
    }
    __syncthreads();
    const int mrow = lane & 15;
    v16h af[2], bf[4];
#pragma unroll
    for (int fm = 0; fm < 2; ++fm)
      af[fm] = load_a_frag(&As[(wm + fm * 16 + mrow) * LDT], lane);
#pragma unroll
    for (int fn = 0; fn < 4; ++fn)
      bf[fn] = load_b_frag(&Bs[(wn + fn * 16 + mrow) * LDT], lane);
#pragma unroll
    for (int fm = 0; fm < 2; ++fm)
#pragma unroll
      for (int fn = 0; fn < 4; ++fn)
        acc[fm][fn] = __builtin_amdgcn_wmma_f32_16x16x32_f16(
            false, af[fm], false, bf[fn], (short)0, acc[fm][fn], false, false);
  }
  const int mr = (lane & 16) ? 8 : 0, nc = lane & 15;
#pragma unroll
  for (int fm = 0; fm < 2; ++fm)
#pragma unroll
    for (int fn = 0; fn < 4; ++fn) {
      int n = n0 + wn + fn * 16 + nc;
#pragma unroll
      for (int r = 0; r < 8; ++r) {
        int m = m0 + wm + fm * 16 + mr + r;
        st[(bh * S_ + m) * S_ + n] = acc[fm][fn][r] * 0.125f;
      }
    }
}

// ---------------------------------------------------------------------------
// Masked softmax over rows of scores_t; masked entries -> exact 0
// ---------------------------------------------------------------------------
__global__ __launch_bounds__(256) void softmax_rows(const float* __restrict__ st,
                                                    float* __restrict__ attn,
                                                    const int* __restrict__ lens) {
  const int row = blockIdx.x;           // (b*H + h)*S + kk
  const int b = row >> 13, kk = row & 511;
  const int pqn = lens[b], pkn = lens[8 + b];
  const float* src = st + (long)row * S_;
  float* dst = attn + (long)row * S_;
  const int t = threadIdx.x, wid = t >> 5, lane = t & 31;
  const bool rowvalid = kk < pkn;
  float x0 = src[t], x1 = src[t + 256];
  bool v0 = rowvalid && (t < pqn);
  bool v1 = rowvalid && (t + 256 < pqn);
  float m = fmaxf(v0 ? x0 : -1e30f, v1 ? x1 : -1e30f);
#pragma unroll
  for (int o = 16; o; o >>= 1) m = fmaxf(m, __shfl_xor(m, o, 32));
  __shared__ float redm[8], reds[8];
  if (lane == 0) redm[wid] = m;
  __syncthreads();
  float mm = redm[0];
#pragma unroll
  for (int i = 1; i < 8; ++i) mm = fmaxf(mm, redm[i]);
  float e0 = v0 ? __expf(x0 - mm) : 0.f;
  float e1 = v1 ? __expf(x1 - mm) : 0.f;
  float s = e0 + e1;
#pragma unroll
  for (int o = 16; o; o >>= 1) s += __shfl_xor(s, o, 32);
  if (lane == 0) reds[wid] = s;
  __syncthreads();
  float ss = 0.f;
#pragma unroll
  for (int i = 0; i < 8; ++i) ss += reds[i];
  float inv = (ss > 0.f) ? 1.f / ss : 0.f;
  dst[t] = e0 * inv;
  dst[t + 256] = e1 * inv;
}

// ---------------------------------------------------------------------------
// context[bh][kk][d] = sum_qq attn[bh][kk][qq] * v1[b][qq][h*64+d]
//   A is fp32 (convert at staging); B needs transpose -> manual staging.
// ---------------------------------------------------------------------------
__global__ __launch_bounds__(256) void gemm_ctx(const float* __restrict__ attnf,
                                                const _Float16* __restrict__ v1p,
                                                float* __restrict__ ctxf,
                                                _Float16* __restrict__ ctx16) {
  __shared__ __attribute__((aligned(16))) _Float16 As[128 * LDT];
  __shared__ __attribute__((aligned(16))) _Float16 Bs[64 * LDT];
  const int m0 = blockIdx.x * 128, bh = blockIdx.z;
  const int b = bh >> 4, hoff = (bh & 15) * DH_;
  const int tid = threadIdx.x, wid = tid >> 5, lane = tid & 31;
  const int wm = (wid & 3) * 32, wn = (wid >> 2) * 32;
  v8f acc[2][2];
#pragma unroll
  for (int i = 0; i < 2; ++i)
#pragma unroll
    for (int j = 0; j < 2; ++j)
#pragma unroll
      for (int r = 0; r < 8; ++r) acc[i][j][r] = 0.f;

  for (int k0 = 0; k0 < S_; k0 += 32) {
    __syncthreads();
#pragma unroll
    for (int i = 0; i < 16; ++i) {            // attn fp32 -> f16 in LDS
      int e = tid + 256 * i, r = e >> 5, c = e & 31;
      As[r * LDT + c] = (_Float16)attnf[(bh * S_ + m0 + r) * S_ + k0 + c];
    }
#pragma unroll
    for (int i = 0; i < 8; ++i) {             // Bs[d][kd] = v1[k0+kd][d] (transpose)
      int e = tid + 256 * i, d = e & 63, kd = e >> 6;
      Bs[d * LDT + kd] = v1p[(b * S_ + k0 + kd) * HID_ + hoff + d];
    }
    __syncthreads();
    const int mrow = lane & 15;
    v16h af[2], bf[2];
#pragma unroll
    for (int fm = 0; fm < 2; ++fm)
      af[fm] = load_a_frag(&As[(wm + fm * 16 + mrow) * LDT], lane);
#pragma unroll
    for (int fn = 0; fn < 2; ++fn)
      bf[fn] = load_b_frag(&Bs[(wn + fn * 16 + mrow) * LDT], lane);
#pragma unroll
    for (int fm = 0; fm < 2; ++fm)
#pragma unroll
      for (int fn = 0; fn < 2; ++fn)
        acc[fm][fn] = __builtin_amdgcn_wmma_f32_16x16x32_f16(
            false, af[fm], false, bf[fn], (short)0, acc[fm][fn], false, false);
  }
  const int mr = (lane & 16) ? 8 : 0, nc = lane & 15;
#pragma unroll
  for (int fm = 0; fm < 2; ++fm)
#pragma unroll
    for (int fn = 0; fn < 2; ++fn) {
      int n = wn + fn * 16 + nc;
#pragma unroll
      for (int r = 0; r < 8; ++r) {
        int m = m0 + wm + fm * 16 + mr + r;
        float v = acc[fm][fn][r];
        int idx = (bh * S_ + m) * DH_ + n;
        ctxf[idx] = v;
        ctx16[idx] = (_Float16)v;
      }
    }
}

// ---------------------------------------------------------------------------
// context2[m][n] = sum_j concat(ctx, vt)[m][j] * Wcat[j][n] + bcat[n]
//   A tiles are f16 row-contiguous (constant row stride per k-step) -> TDM.
// ---------------------------------------------------------------------------
__global__ __launch_bounds__(256) void gemm_cat(const _Float16* __restrict__ ctx16,
                                                const _Float16* __restrict__ vtp,
                                                const float* __restrict__ Wcat,
                                                const float* __restrict__ bcat,
                                                float* __restrict__ out2) {
  __shared__ __attribute__((aligned(16))) _Float16 As[128 * LDT];
  __shared__ __attribute__((aligned(16))) _Float16 Bs[64 * LDT];
  const int m0 = blockIdx.x * 128;
  const int tid = threadIdx.x, wid = tid >> 5, lane = tid & 31;
  const int wm = (wid & 3) * 32, wn = (wid >> 2) * 32;
  v8f acc[2][2];
#pragma unroll
  for (int i = 0; i < 2; ++i)
#pragma unroll
    for (int j = 0; j < 2; ++j)
#pragma unroll
      for (int r = 0; r < 8; ++r) acc[i][j][r] = 0.f;

  for (int k0 = 0; k0 < 128; k0 += 32) {
    __syncthreads();
#pragma unroll
    for (int i = 0; i < 8; ++i) {             // Bs[d][kd] = Wcat[k0+kd][d] (fp32->f16)
      int e = tid + 256 * i, d = e & 63, kd = e >> 6;
      Bs[d * LDT + kd] = (_Float16)Wcat[(k0 + kd) * DH_ + d];
    }
    if (tid < 32) {                           // A tile via TDM (f16 source)
      const _Float16* abase;
      unsigned long long stride;
      if (k0 < 64) {
        abase = ctx16 + (size_t)m0 * DH_ + k0;
        stride = DH_;
      } else {
        int bb = m0 >> 13, hh = (m0 >> 9) & 15, ss = m0 & 511;
        abase = vtp + (size_t)(bb * S_ + ss) * HID_ + hh * DH_ + (k0 - 64);
        stride = HID_;
      }
      tdm_load_tile_f16(abase, (unsigned)(size_t)&As[0], 128, stride);
      __builtin_amdgcn_s_wait_tensorcnt(0);
    }
    __syncthreads();
    const int mrow = lane & 15;
    v16h af[2], bf[2];
#pragma unroll
    for (int fm = 0; fm < 2; ++fm)
      af[fm] = load_a_frag(&As[(wm + fm * 16 + mrow) * LDT], lane);
#pragma unroll
    for (int fn = 0; fn < 2; ++fn)
      bf[fn] = load_b_frag(&Bs[(wn + fn * 16 + mrow) * LDT], lane);
#pragma unroll
    for (int fm = 0; fm < 2; ++fm)
#pragma unroll
      for (int fn = 0; fn < 2; ++fn)
        acc[fm][fn] = __builtin_amdgcn_wmma_f32_16x16x32_f16(
            false, af[fm], false, bf[fn], (short)0, acc[fm][fn], false, false);
  }
  const int mr = (lane & 16) ? 8 : 0, nc = lane & 15;
#pragma unroll
  for (int fm = 0; fm < 2; ++fm)
#pragma unroll
    for (int fn = 0; fn < 2; ++fn) {
      int n = wn + fn * 16 + nc;
      float bb = bcat[n];
#pragma unroll
      for (int r = 0; r < 8; ++r) {
        int m = m0 + wm + fm * 16 + mr + r;
        out2[m * DH_ + n] = acc[fm][fn][r] + bb;
      }
    }
}

// ---------------------------------------------------------------------------
extern "C" void kernel_launch(void* const* d_in, const int* in_sizes, int n_in,
                              void* d_out, int out_size, void* d_ws, size_t ws_size,
                              hipStream_t stream) {
  (void)in_sizes; (void)n_in; (void)out_size; (void)ws_size;
  const int*   pq       = (const int*)d_in[0];
  const int*   pk       = (const int*)d_in[1];
  const float* hid_last = (const float*)d_in[6];
  const float* hid_cur  = (const float*)d_in[7];
  const float* W_v1 = (const float*)d_in[11]; const float* b_v1 = (const float*)d_in[12];
  const float* W_v  = (const float*)d_in[13]; const float* b_v  = (const float*)d_in[14];
  const float* W_q  = (const float*)d_in[15]; const float* b_q  = (const float*)d_in[16];
  const float* W_k  = (const float*)d_in[17]; const float* b_k  = (const float*)d_in[18];
  const float* W_cat= (const float*)d_in[19]; const float* b_cat= (const float*)d_in[20];

  float* out      = (float*)d_out;
  float* attn_out = out;                 // [B,H,S,S]  16,777,216
  float* st_out   = out + 16777216;      // scores_t   16,777,216
  float* c2_out   = out + 33554432;      // context2    4,194,304
  float* ctx_out  = out + 37748736;      // context     4,194,304

  char* ws = (char*)d_ws;
  _Float16* v1p   = (_Float16*)(ws + 0);
  _Float16* vtp   = (_Float16*)(ws + 8388608);
  _Float16* qp    = (_Float16*)(ws + 16777216);
  _Float16* kpj   = (_Float16*)(ws + 25165824);
  _Float16* ctx16 = (_Float16*)(ws + 33554432);
  int*      lens  = (int*)(ws + 41943040);

  lens_kernel<<<1, 256, 0, stream>>>(pq, pk, lens);

  dim3 gp(32, 8);   // M/128 x N/128
  gemm_proj<<<gp, 256, 0, stream>>>(hid_last, W_v1, b_v1, v1p);
  gemm_proj<<<gp, 256, 0, stream>>>(hid_cur,  W_v,  b_v,  vtp);
  gemm_proj<<<gp, 256, 0, stream>>>(hid_last, W_q,  b_q,  qp);
  gemm_proj<<<gp, 256, 0, stream>>>(hid_cur,  W_k,  b_k,  kpj);

  dim3 gs(4, 4, 128);
  gemm_scores<<<gs, 256, 0, stream>>>(kpj, qp, st_out);

  softmax_rows<<<B_ * H_ * S_, 256, 0, stream>>>(st_out, attn_out, lens);

  dim3 gc(4, 1, 128);
  gemm_ctx<<<gc, 256, 0, stream>>>(attn_out, v1p, ctx_out, ctx16);

  gemm_cat<<<512, 256, 0, stream>>>(ctx16, vtp, W_cat, b_cat, c2_out);
}
